// AccMcCnnInfer_58506044506221
// MI455X (gfx1250) — matrix operand order repaired and checked by reading, not compiled, not run on hardware
//
#include <hip/hip_runtime.h>
#include <math.h>

// ---------------------------------------------------------------------------
// MC-CNN stereo inference for gfx1250 (MI455X).
// WMMA f16 GEMMs + async global->LDS staging (ASYNCcnt path).
// ---------------------------------------------------------------------------

typedef __attribute__((ext_vector_type(16))) _Float16 v16h;
typedef __attribute__((ext_vector_type(8)))  float    v8f;

union V16H {
  v16h  v;
  uint4 q[2];
};

#define NFM   112   // conv feature maps
#define CPAD  128   // padded channel stride (zero-filled 112..127)
#define W0    394   // input image width
#define HF    192   // final feature height
#define WF    384   // final feature width
#define NDISP 32    // disp_max - disp_min + 1 (from setup_inputs)
#define MLPH  384
#define CIN2  224   // concat channels into MLP layer 1

// Async copy of one 16-byte chunk from global to LDS (gfx1250, ASYNCcnt).
__device__ __forceinline__ void async_copy16(unsigned lds_addr, const void* gptr) {
  unsigned long long ga = (unsigned long long)(size_t)gptr;
  asm volatile("global_load_async_to_lds_b128 %0, %1, off"
               :: "v"(lds_addr), "v"(ga) : "memory");
}
__device__ __forceinline__ void async_wait0() {
  asm volatile("s_wait_asynccnt 0x0" ::: "memory");
}

// ---------------------------------------------------------------------------
// Weight preparation
// ---------------------------------------------------------------------------

// cw (112,112,3,3) OIHW fp32  ->  dst[(kk*112 + co)*128 + ci] f16, zero-padded
__global__ void prep_convw(const float* __restrict__ src, _Float16* __restrict__ dst) {
  int idx = blockIdx.x * blockDim.x + threadIdx.x;
  const int total = 9 * NFM * CPAD;
  if (idx >= total) return;
  int ci = idx & (CPAD - 1);
  int t  = idx >> 7;          // kk*112 + co
  int co = t % NFM;
  int kk = t / NFM;           // ky*3 + kx
  _Float16 v = (_Float16)0.0f;
  if (ci < NFM) v = (_Float16)src[(co * NFM + ci) * 9 + kk];
  dst[idx] = v;
}

// fw (din,dout) fp32 -> dst[o*din + i] f16 (transposed, K contiguous per column)
__global__ void prep_fcw(const float* __restrict__ src, _Float16* __restrict__ dst,
                         int din, int dout) {
  int idx = blockIdx.x * blockDim.x + threadIdx.x;
  if (idx >= din * dout) return;
  int i = idx % din;
  int o = idx / din;
  dst[o * din + i] = (_Float16)src[i * dout + o];
}

// ---------------------------------------------------------------------------
// Conv layer 1: 1 -> 112 channels, K=9, trivial VALU work.
// ---------------------------------------------------------------------------
__global__ void conv1_kernel(const float* __restrict__ img, const float* __restrict__ w,
                             const float* __restrict__ b, _Float16* __restrict__ out,
                             int Hout, int Wout, int Win) {
  int idx = blockIdx.x * blockDim.x + threadIdx.x;
  int total = Hout * Wout * CPAD;
  if (idx >= total) return;
  int c   = idx & (CPAD - 1);
  int pix = idx >> 7;
  int x   = pix % Wout;
  int y   = pix / Wout;
  _Float16 r = (_Float16)0.0f;
  if (c < NFM) {
    float s = b[c];
#pragma unroll
    for (int ky = 0; ky < 3; ++ky)
#pragma unroll
      for (int kx = 0; kx < 3; ++kx)
        s += img[(y + ky) * Win + (x + kx)] * w[c * 9 + ky * 3 + kx];
    r = (_Float16)(s > 0.0f ? s : 0.0f);
  }
  out[idx] = r;
}

// ---------------------------------------------------------------------------
// Conv layers 2..5 via WMMA. Block = 7 waves; wave w owns cout tile w*16.
// Input tile (3 rows x 18 cols x 128 ch) staged once in LDS via async copy.
// ---------------------------------------------------------------------------
__global__ __launch_bounds__(224) void convN_kernel(
    const _Float16* __restrict__ in, const _Float16* __restrict__ w,
    const float* __restrict__ bias, _Float16* __restrict__ out,
    int Win, int Wout) {
  __shared__ __align__(16) _Float16 sT[3 * 18 * CPAD];   // 13824 B

  const int wv   = threadIdx.x >> 5;        // 0..6  -> cout tile
  const int lane = threadIdx.x & 31;
  const int row  = blockIdx.y;              // output row
  const int col0 = blockIdx.x * 16;
  const int mrow = lane & 15;
  const int asel = (lane < 16) ? 0 : 8;     // A: interleaved 8-half groups
  const int bsel = (lane < 16) ? 0 : 8;     // B: interleaved 8-half groups

  // ---- Stage input patch into LDS (async, 16B chunks) --------------------
  {
    const unsigned ldsbase = (unsigned)(size_t)(&sT[0]);
    for (int e = threadIdx.x; e < 3 * 18 * 16; e += 224) {
      int ch0 = (e & 15) * 8;               // half index 0..120
      int cc  = (e >> 4) % 18;              // local col
      int rr  = (e >> 4) / 18;              // local row
      int sc  = col0 + cc;
      if (sc > Win - 1) sc = Win - 1;       // clamp (writes masked later)
      const _Float16* gp = in + ((size_t)(row + rr) * Win + sc) * CPAD + ch0;
      unsigned lo = ldsbase + (unsigned)(((rr * 18 + cc) * CPAD + ch0) * 2);
      async_copy16(lo, gp);
    }
    async_wait0();
    __syncthreads();
  }

  // ---- GEMM: M=16 pixels, N=16 couts (this wave), K=9*128 ----------------
  v8f acc = {};
  for (int ky = 0; ky < 3; ++ky) {
    for (int kx = 0; kx < 3; ++kx) {
      const _Float16* ap = &sT[(ky * 18 + mrow + kx) * CPAD + asel];
      const _Float16* wp = w + ((size_t)(ky * 3 + kx) * NFM + wv * 16 + mrow) * CPAD + bsel;
#pragma unroll
      for (int kc = 0; kc < CPAD; kc += 32) {
        V16H a, b;
        a.q[0] = *(const uint4*)(ap + kc);
        a.q[1] = *(const uint4*)(ap + kc + 16);
        b.q[0] = *(const uint4*)(wp + kc);
        b.q[1] = *(const uint4*)(wp + kc + 16);
        acc = __builtin_amdgcn_wmma_f32_16x16x32_f16(
            false, a.v, false, b.v, (short)0, acc, false, false);
      }
    }
  }

  // Writeback C tile: lane<16: N=lane, M=r; lane>=16: N=lane-16, M=r+8
  const int n     = wv * 16 + (lane & 15);
  const int mbase = (lane < 16) ? 0 : 8;
  const float bv  = bias[n];
#pragma unroll
  for (int r = 0; r < 8; ++r) {
    int px = col0 + mbase + r;
    if (px < Wout) {
      float v = acc[r] + bv;
      v = v > 0.0f ? v : 0.0f;
      out[((size_t)row * Wout + px) * CPAD + n] = (_Float16)v;
    }
  }
  // Zero padded channels 112..127 for this tile's pixels.
  if (threadIdx.x < 16) {
    int px = col0 + (int)threadIdx.x;
    if (px < Wout) {
      uint4 z = {0, 0, 0, 0};
      uint4* q = (uint4*)(out + ((size_t)row * Wout + px) * CPAD + NFM);
      q[0] = z;
      q[1] = z;
    }
  }
}

// ---------------------------------------------------------------------------
// Fused MLP: one block = 32 pixels (one row segment) x one disparity.
// 8 waves; each wave owns 3 N-tiles x 2 M-tiles per FC layer (B reused 2x).
// ---------------------------------------------------------------------------
template <int K>
__device__ __forceinline__ void fc_layer(const _Float16* sIn,
                                         const _Float16* __restrict__ wT,
                                         const float* __restrict__ bias,
                                         _Float16* sOut) {
  const int wv   = threadIdx.x >> 5;   // 0..7
  const int lane = threadIdx.x & 31;
  const int mrow = lane & 15;
  const int asel = (lane < 16) ? 0 : 8;
  const int bsel = (lane < 16) ? 0 : 8;
  const int nb   = wv * 48;

  v8f acc[2][3] = {};
  for (int k0 = 0; k0 < K; k0 += 32) {
    V16H a0, a1;
    const _Float16* ap = sIn + mrow * K + k0 + asel;
    a0.q[0] = *(const uint4*)(ap);
    a0.q[1] = *(const uint4*)(ap + 16);
    ap += 16 * K;
    a1.q[0] = *(const uint4*)(ap);
    a1.q[1] = *(const uint4*)(ap + 16);
    const _Float16* wp = wT + (size_t)(nb + mrow) * K + k0 + bsel;
#pragma unroll
    for (int t = 0; t < 3; ++t) {
      V16H b;
      b.q[0] = *(const uint4*)(wp);
      b.q[1] = *(const uint4*)(wp + 16);
      acc[0][t] = __builtin_amdgcn_wmma_f32_16x16x32_f16(
          false, a0.v, false, b.v, (short)0, acc[0][t], false, false);
      acc[1][t] = __builtin_amdgcn_wmma_f32_16x16x32_f16(
          false, a1.v, false, b.v, (short)0, acc[1][t], false, false);
      wp += 16 * K;
    }
  }

  const int nlo   = lane & 15;
  const int mbase = (lane < 16) ? 0 : 8;
#pragma unroll
  for (int t = 0; t < 3; ++t) {
    int n    = nb + t * 16 + nlo;
    float bv = bias[n];
#pragma unroll
    for (int mt = 0; mt < 2; ++mt)
#pragma unroll
      for (int r = 0; r < 8; ++r) {
        float v = acc[mt][t][r] + bv;
        v = v > 0.0f ? v : 0.0f;
        sOut[(mt * 16 + mbase + r) * MLPH + n] = (_Float16)v;
      }
  }
}

__global__ __launch_bounds__(256) void mlp_kernel(
    const _Float16* __restrict__ lf, const _Float16* __restrict__ rf,
    const _Float16* __restrict__ w1t, const float* __restrict__ fb1,
    const _Float16* __restrict__ w2t, const float* __restrict__ fb2,
    const _Float16* __restrict__ w3t, const float* __restrict__ fb3,
    const float* __restrict__ fw4, const float* __restrict__ fb4,
    const int* __restrict__ dminp, float* __restrict__ out) {
  __shared__ __align__(16) _Float16 sA[32 * CIN2];   // 14336 B
  __shared__ __align__(16) _Float16 sB[32 * MLPH];   // 24576 B
  __shared__ __align__(16) _Float16 sC[32 * MLPH];   // 24576 B

  const int row  = blockIdx.y;
  const int col0 = blockIdx.x * 32;
  const int dIdx = blockIdx.z;
  const int d    = dminp[0] + dIdx;

  // Assemble concat tile via async gather: ch 0..111 = lf, 112..223 = rf shifted.
  {
    const unsigned sAbase = (unsigned)(size_t)(&sA[0]);
    for (int e = threadIdx.x; e < 32 * 28; e += 256) {  // 28 x 16B chunks / pixel
      int m  = e / 28;
      int ch = (e % 28) * 8;                            // half index 0..216
      int px = col0 + m;
      const _Float16* gp;
      if (ch < NFM) {
        gp = lf + ((size_t)row * WF + px) * CPAD + ch;
      } else {
        int sx = px + d;
        sx = sx < 0 ? 0 : (sx > WF - 1 ? WF - 1 : sx);
        gp = rf + ((size_t)row * WF + sx) * CPAD + (ch - NFM);
      }
      unsigned lo = sAbase + (unsigned)((m * CIN2 + ch) * 2);
      async_copy16(lo, gp);
    }
    async_wait0();
    __syncthreads();
  }

  fc_layer<CIN2>(sA, w1t, fb1, sB);   // 224 -> 384
  __syncthreads();
  fc_layer<MLPH>(sB, w2t, fb2, sC);   // 384 -> 384
  __syncthreads();
  fc_layer<MLPH>(sC, w3t, fb3, sB);   // 384 -> 384 (h3 in sB)
  __syncthreads();

  // Layer 4: 384 -> 1 dot + sigmoid. 16 lanes cooperate per pixel; 2 pixels/thread.
  const int j  = threadIdx.x & 15;
  const int mb = threadIdx.x >> 4;    // 0..15
#pragma unroll
  for (int mt = 0; mt < 2; ++mt) {
    int m = mb + mt * 16;
    float part = 0.0f;
    for (int c = j; c < MLPH; c += 16)
      part += (float)sB[m * MLPH + c] * fw4[c];
    part += __shfl_xor(part, 8);
    part += __shfl_xor(part, 4);
    part += __shfl_xor(part, 2);
    part += __shfl_xor(part, 1);
    if (j == 0) {
      float x     = part + fb4[0];
      float score = 1.0f / (1.0f + __expf(-x));
      int px      = col0 + m;
      int sx      = px + d;
      bool valid  = (sx >= 0) && (sx < WF);
      out[((size_t)row * WF + px) * NDISP + dIdx] =
          valid ? -score : __int_as_float(0x7FC00000);
    }
  }
}

// ---------------------------------------------------------------------------
// Host launcher
// ---------------------------------------------------------------------------
extern "C" void kernel_launch(void* const* d_in, const int* in_sizes, int n_in,
                              void* d_out, int out_size, void* d_ws, size_t ws_size,
                              hipStream_t stream) {
  (void)in_sizes; (void)n_in; (void)out_size; (void)ws_size;

  const float* left  = (const float*)d_in[0];
  const float* right = (const float*)d_in[1];
  const float* cw1   = (const float*)d_in[2];
  const float* cb1   = (const float*)d_in[3];
  const float* cw[4] = {(const float*)d_in[4], (const float*)d_in[6],
                        (const float*)d_in[8], (const float*)d_in[10]};
  const float* cb[4] = {(const float*)d_in[5], (const float*)d_in[7],
                        (const float*)d_in[9], (const float*)d_in[11]};
  const float* fw1 = (const float*)d_in[12];
  const float* fb1 = (const float*)d_in[13];
  const float* fw2 = (const float*)d_in[14];
  const float* fb2 = (const float*)d_in[15];
  const float* fw3 = (const float*)d_in[16];
  const float* fb3 = (const float*)d_in[17];
  const float* fw4 = (const float*)d_in[18];
  const float* fb4 = (const float*)d_in[19];
  const int*   dmin = (const int*)d_in[20];
  float* out = (float*)d_out;

  // Carve workspace (~76 MB total).
  char*  ws  = (char*)d_ws;
  size_t off = 0;
  auto take = [&](size_t bytes) -> char* {
    char* p = ws + off;
    off += (bytes + 255) & ~(size_t)255;
    return p;
  };
  _Float16* wc[4];
  for (int l = 0; l < 4; ++l)
    wc[l] = (_Float16*)take((size_t)9 * NFM * CPAD * sizeof(_Float16));
  _Float16* w1t  = (_Float16*)take((size_t)MLPH * CIN2 * sizeof(_Float16));
  _Float16* w2t  = (_Float16*)take((size_t)MLPH * MLPH * sizeof(_Float16));
  _Float16* w3t  = (_Float16*)take((size_t)MLPH * MLPH * sizeof(_Float16));
  _Float16* bufA = (_Float16*)take((size_t)200 * 392 * CPAD * sizeof(_Float16));
  _Float16* bufB = (_Float16*)take((size_t)200 * 392 * CPAD * sizeof(_Float16));
  _Float16* lfb  = (_Float16*)take((size_t)HF * WF * CPAD * sizeof(_Float16));
  _Float16* rfb  = (_Float16*)take((size_t)HF * WF * CPAD * sizeof(_Float16));

  // Weight prep (cheap; re-run every call for determinism).
  {
    int tot = 9 * NFM * CPAD;
    for (int l = 0; l < 4; ++l)
      prep_convw<<<(tot + 255) / 256, 256, 0, stream>>>(cw[l], wc[l]);
    prep_fcw<<<(CIN2 * MLPH + 255) / 256, 256, 0, stream>>>(fw1, w1t, CIN2, MLPH);
    prep_fcw<<<(MLPH * MLPH + 255) / 256, 256, 0, stream>>>(fw2, w2t, MLPH, MLPH);
    prep_fcw<<<(MLPH * MLPH + 255) / 256, 256, 0, stream>>>(fw3, w3t, MLPH, MLPH);
  }

  // Conv towers for both images.
  const int Ws[5] = {392, 390, 388, 386, 384};
  const int Hs[5] = {200, 198, 196, 194, 192};
  for (int img = 0; img < 2; ++img) {
    const float* src = (img == 0) ? left : right;
    _Float16* dstF   = (img == 0) ? lfb : rfb;
    int tot1 = Hs[0] * Ws[0] * CPAD;
    conv1_kernel<<<(tot1 + 255) / 256, 256, 0, stream>>>(src, cw1, cb1, bufA,
                                                         Hs[0], Ws[0], W0);
    _Float16* pin  = bufA;
    _Float16* pout = bufB;
    for (int l = 0; l < 4; ++l) {
      _Float16* o = (l == 3) ? dstF : pout;
      dim3 g((Ws[l + 1] + 15) / 16, Hs[l + 1]);
      convN_kernel<<<g, 224, 0, stream>>>(pin, wc[l], cb[l], o, Ws[l], Ws[l + 1]);
      _Float16* t = pin; pin = pout; pout = t;
    }
  }

  // Fused MLP cost volume: (12 col tiles of 32 px, 192 rows, 32 disparities).
  dim3 gm(WF / 32, HF, NDISP);
  mlp_kernel<<<gm, 256, 0, stream>>>(lfb, rfb, w1t, fb1, w2t, fb2, w3t, fb3,
                                     fw4, fb4, dmin, out);
}
